// VNet_77189152244218
// MI455X (gfx1250) — compile-verified, matrix operand order checked by reading, not run on hardware
//
#include <hip/hip_runtime.h>
#include <math.h>

typedef _Float16 h16;
typedef __attribute__((ext_vector_type(16))) _Float16 v16h;
typedef __attribute__((ext_vector_type(8)))  _Float16 v8h;
typedef __attribute__((ext_vector_type(8)))  float    v8f;

// ---------------- problem constants ----------------
enum { B = 16, N = 4096, S1 = 512, S2 = 128, MM = 32 };

// ---------------- descriptors ----------------
struct Layer {
  const h16*   W;      // (Cout, CinPad) f16, zero padded
  const float* bias;   // (Cout)
  const float* acoef;  // per-channel BN scale  (gamma * rsqrt(var+eps))
  const float* ccoef;  // per-channel BN shift  (beta - mean*acoef)
  int Cin, CinPad, Cout;
};

struct FArgs {
  int mode;                 // 0 = ball-group (xyz/points/new_xyz), 1 = direct rows
  const float* xyz;         // (B,Nn,3)
  const float* points;      // (B,Nn,Cpts) or null
  const float* newxyz;      // (G,3)  G = B*Ss
  const float* rows;        // mode1 rows (G*K, Cin0)
  int Nn, Ss, Kk, Cpts, Cin0;
  float r2;
  Layer L[3];
  int nl;                   // layers computed this pass
  int apply_last;           // 1: apply bn+relu on last layer and emit output
  float* stats;             // [2*CoutLast] sum / sumsq (global atomics)
  int do_pool;              // max-pool over K within group
  float* out;
  int out_stride, out_off;
};

struct ConvP {
  const float *b, *g, *be;
  int Cin, Cout, CinPad;
  h16* Wp;
};

// ---------------- small utility kernels ----------------
__global__ void k_zero(float* p, int n) {
  int i = blockIdx.x * blockDim.x + threadIdx.x;
  if (i < n) p[i] = 0.f;
}

__global__ void k_cvtw(const float* __restrict__ W, h16* __restrict__ out,
                       int Cin, int CinPad, int tot) {
  int i = blockIdx.x * blockDim.x + threadIdx.x;
  if (i >= tot) return;
  int r = i / CinPad, c = i - r * CinPad;
  out[i] = (h16)(c < Cin ? W[(size_t)r * Cin + c] : 0.f);
}

// (B,C,Nn) -> (B*Nn, C)
__global__ void k_bcn_to_bnc(const float* __restrict__ in, float* __restrict__ out,
                             int Bb, int C, int Nn) {
  int i = blockIdx.x * blockDim.x + threadIdx.x;
  int tot = Bb * C * Nn;
  if (i >= tot) return;
  int n = i % Nn, c = (i / Nn) % C, b = i / (Nn * C);
  out[((size_t)b * Nn + n) * C + c] = in[i];
}

__global__ void k_copy3(const float* __restrict__ nx, float* __restrict__ l2cat, int G) {
  int i = blockIdx.x * blockDim.x + threadIdx.x;
  if (i >= G * 3) return;
  int g = i / 3, c = i - g * 3;
  l2cat[(size_t)g * 323 + c] = nx[i];
}

__global__ void k_fuse(const float* __restrict__ fs, const float* __restrict__ fo,
                       const float* __restrict__ z, float* __restrict__ out) {
  int i = blockIdx.x * blockDim.x + threadIdx.x;
  int tot = B * MM * 320;
  if (i >= tot) return;
  int r = i / 320, c = i - r * 320, b = r / MM;
  float v;
  if (c < 128)      v = fs[b * 128 + c];
  else if (c < 256) v = fo[b * 128 + (c - 128)];
  else              v = z[(size_t)r * 64 + (c - 256)];
  out[i] = v;
}

__global__ void k_finalize(const float* __restrict__ stats,
                           const float* __restrict__ gamma, const float* __restrict__ beta,
                           float* __restrict__ a, float* __restrict__ c, int C, float invR) {
  int n = blockIdx.x * blockDim.x + threadIdx.x;
  if (n >= C) return;
  float m  = stats[n] * invR;
  float v  = stats[C + n] * invR - m * m;
  float iv = rsqrtf(v + 1e-5f);
  float aa = gamma[n] * iv;
  a[n] = aa;
  c[n] = beta[n] - m * aa;
}

// ---------------- FPS: one block per batch, xyz cached in LDS ----------------
__global__ __launch_bounds__(256) void k_fps(const float* __restrict__ xyz,
                                             float* __restrict__ nxyz, int Nn, int Ss) {
  __shared__ float sx[4096], sy[4096], sz[4096];
  __shared__ float rmax[256];
  __shared__ int   rarg[256];
  int b = blockIdx.x, t = threadIdx.x;
  for (int i = t; i < Nn; i += 256) {
    const float* q = &xyz[((size_t)b * Nn + i) * 3];
    sx[i] = q[0]; sy[i] = q[1]; sz[i] = q[2];
  }
  __syncthreads();
  const int PPT = Nn / 256;          // 16 for N=4096, 2 for N=512
  float dist[16];
  for (int j = 0; j < PPT; j++) dist[j] = 1e10f;
  int far = 0;
  for (int it = 0; it < Ss; it++) {
    if (t == 0) {
      float* o = &nxyz[((size_t)b * Ss + it) * 3];
      o[0] = sx[far]; o[1] = sy[far]; o[2] = sz[far];
    }
    float cx = sx[far], cy = sy[far], cz = sz[far];
    float lm = -1.f; int la = 0;
    for (int j = 0; j < PPT; j++) {
      int p = j * 256 + t;
      float dx = sx[p] - cx, dy = sy[p] - cy, dz = sz[p] - cz;
      float d = dx * dx + dy * dy + dz * dz;
      float dd = fminf(dist[j], d);
      dist[j] = dd;
      if (dd > lm) { lm = dd; la = p; }
    }
    rmax[t] = lm; rarg[t] = la;
    __syncthreads();
    for (int s = 128; s > 0; s >>= 1) {
      if (t < s && rmax[t + s] > rmax[t]) { rmax[t] = rmax[t + s]; rarg[t] = rarg[t + s]; }
      __syncthreads();
    }
    far = rarg[0];
    __syncthreads();
  }
}

// ---------------- fused grouped-MLP kernel (WMMA) ----------------
// One workgroup (4 wave32) per group. Activations live in LDS as f16.
template <int RMAX, int CA, int CB>
__global__ __launch_bounds__(128) void k_fused(FArgs A) {
  __shared__ h16 bufA[RMAX * CA];
  __shared__ h16 bufB[RMAX * CB];
  __shared__ int sIdx[RMAX];
  __shared__ float sPool[512];
  __shared__ int sCnt;

  const int tid = threadIdx.x, lane = tid & 31, wv = tid >> 5;
  const int g = blockIdx.x, K = A.Kk;
  int b = 0;

  for (int i = tid; i < RMAX * CA; i += 128) bufA[i] = (h16)0.f;
  __syncthreads();

  // ---- stage input A0 (raw, no normalization — matches reference) ----
  if (A.mode == 0) {
    b = g / A.Ss;
    const float cx = A.newxyz[g * 3 + 0], cy = A.newxyz[g * 3 + 1], cz = A.newxyz[g * 3 + 2];
    if (tid < 32) {                      // wave-0 ballot ball-query, preserves index order
      unsigned found = 0u;
      for (int base = 0; base < A.Nn; base += 32) {
        int p = base + lane;
        bool in = false;
        if (p < A.Nn) {
          const float* q = &A.xyz[((size_t)b * A.Nn + p) * 3];
          float dx = q[0] - cx, dy = q[1] - cy, dz = q[2] - cz;
          in = (dx * dx + dy * dy + dz * dz) <= A.r2;
        }
        unsigned m = (unsigned)__ballot(in);          // wave32: low 32 bits
        if (in) {
          unsigned pos = found + (unsigned)__popc(m & ((1u << lane) - 1u));
          if (pos < (unsigned)K) sIdx[pos] = p;
        }
        found += (unsigned)__popc(m);
        if (found >= (unsigned)K) break;              // uniform
      }
      if (lane == 0) sCnt = (int)(found < (unsigned)K ? found : (unsigned)K);
    }
    __syncthreads();
    int cnt = sCnt;
    int fill = (cnt > 0) ? sIdx[0] : 0;               // pad with first (reference semantics)
    for (int j = tid; j < K; j += 128) if (j >= cnt) sIdx[j] = fill;
    __syncthreads();
    int total = K * A.Cin0;
    for (int i = tid; i < total; i += 128) {
      int j = i / A.Cin0, c = i - j * A.Cin0;
      int p = sIdx[j];
      float v;
      if (c < A.Cpts) v = A.points[((size_t)b * A.Nn + p) * A.Cpts + c];
      else {
        int cc = c - A.Cpts;
        v = A.xyz[((size_t)b * A.Nn + p) * 3 + cc] - A.newxyz[g * 3 + cc];
      }
      bufA[j * CA + c] = (h16)v;
    }
  } else {
    int total = K * A.Cin0;
    for (int i = tid; i < total; i += 128) {
      int j = i / A.Cin0, c = i - j * A.Cin0;
      bufA[j * CA + c] = (h16)A.rows[((size_t)g * K + j) * A.Cin0 + c];
    }
  }
  __syncthreads();

  // ---- layer chain ----
  const h16* act = bufA;
  int actStride = CA;
  const int half = lane >> 4, l15 = lane & 15;

  for (int li = 0; li < A.nl; li++) {
    const Layer L = A.L[li];
    const bool last = (li == A.nl - 1);
    h16* outbuf = (li & 1) ? bufA : bufB;       // ping-pong
    const int outStride = (li & 1) ? CA : CB;
    if (!last) { for (int i = tid; i < RMAX * outStride; i += 128) outbuf[i] = (h16)0.f; }
    if (last && A.apply_last && A.do_pool) { for (int i = tid; i < 512; i += 128) sPool[i] = 0.f; }
    __syncthreads();

    const int TM = K >> 4, TN = L.Cout >> 4, NT = TM * TN, KC = L.CinPad >> 5;
    for (int t = wv; t < NT; t += 4) {          // wave-uniform tile loop (EXEC all ones)
      int tm = t / TN, tn = t - tm * TN;
      v8f acc = {0.f, 0.f, 0.f, 0.f, 0.f, 0.f, 0.f, 0.f};
      const h16* arow = act + (size_t)(tm * 16 + l15) * actStride;
      const h16* brow = L.W + (size_t)(tn * 16 + l15) * L.CinPad;
      for (int kc = 0; kc < KC; kc++) {
        int k0 = kc * 32 + half * 8;            // 16B aligned runs -> ds_load_b128
        v8h alo = *(const v8h*)(arow + k0);
        v8h ahi = *(const v8h*)(arow + k0 + 16);
        v8h blo = *(const v8h*)(brow + k0);
        v8h bhi = *(const v8h*)(brow + k0 + 16);
        v16h av, bv;
#pragma unroll
        for (int e = 0; e < 8; e++) { av[e] = alo[e]; av[e + 8] = ahi[e]; bv[e] = blo[e]; bv[e + 8] = bhi[e]; }
        acc = __builtin_amdgcn_wmma_f32_16x16x32_f16(false, av, false, bv, (short)0, acc, false, false);
      }
      const int n = tn * 16 + l15;              // this lane's output column
      const int r0 = tm * 16 + half * 8;        // first of its 8 rows
      const float bs = L.bias[n];
      if (!last) {                               // intermediate: bn+relu with finalized coefs
        float aa = L.acoef[n], cc = L.ccoef[n];
#pragma unroll
        for (int i = 0; i < 8; i++) {
          float y = fmaxf(aa * (acc[i] + bs) + cc, 0.f);
          outbuf[(size_t)(r0 + i) * outStride + n] = (h16)y;
        }
      } else if (!A.apply_last) {                // stats pass: accumulate sum / sumsq
        float s = 0.f, sq = 0.f;
#pragma unroll
        for (int i = 0; i < 8; i++) { float y = acc[i] + bs; s += y; sq += y * y; }
        atomicAdd(&A.stats[n], s);
        atomicAdd(&A.stats[L.Cout + n], sq);
      } else if (A.do_pool) {                    // output pass with max over K
        float aa = L.acoef[n], cc = L.ccoef[n], mx = 0.f;
#pragma unroll
        for (int i = 0; i < 8; i++) { float y = fmaxf(aa * (acc[i] + bs) + cc, 0.f); mx = fmaxf(mx, y); }
        atomicMax((int*)&sPool[n], __float_as_int(mx));   // post-ReLU >= 0: int order == float order
      } else {                                   // output pass, row output
        float aa = L.acoef[n], cc = L.ccoef[n];
#pragma unroll
        for (int i = 0; i < 8; i++) {
          float y = fmaxf(aa * (acc[i] + bs) + cc, 0.f);
          A.out[((size_t)g * K + r0 + i) * A.out_stride + A.out_off + n] = y;
        }
      }
    }
    if (!last) { __syncthreads(); act = outbuf; actStride = outStride; }
  }

  if (A.apply_last && A.do_pool) {
    __syncthreads();
    int C = A.L[A.nl - 1].Cout;
    for (int n = tid; n < C; n += 128)
      A.out[(size_t)g * A.out_stride + A.out_off + n] = sPool[n];
  }
}

// ---------------- regression head + 6D rotation assembly ----------------
__global__ void k_reg(const float* __restrict__ f, const float* __restrict__ W1,
                      const float* __restrict__ b1, const float* __restrict__ W2,
                      const float* __restrict__ b2, float* __restrict__ out, int R) {
  int r = blockIdx.x * blockDim.x + threadIdx.x;
  if (r >= R) return;
  float y1[32];
  for (int j = 0; j < 32; j++) {
    float s = b1[j];
    for (int c = 0; c < 64; c++) s += f[(size_t)r * 64 + c] * W1[j * 64 + c];
    y1[j] = s;
  }
  float y2[9];
  for (int k = 0; k < 9; k++) {
    float s = b2[k];
    for (int j = 0; j < 32; j++) s += y1[j] * W2[k * 32 + j];
    y2[k] = s;
  }
  float tx = y2[0], ty = y2[1], tz = y2[2];
  float ax = y2[3], ay = y2[4], az = y2[5];
  float n1 = fmaxf(sqrtf(ax * ax + ay * ay + az * az), 1e-12f);
  float b1x = ax / n1, b1y = ay / n1, b1z = az / n1;
  float cx = y2[6], cy = y2[7], cz = y2[8];
  float d = b1x * cx + b1y * cy + b1z * cz;
  float ux = cx - d * b1x, uy = cy - d * b1y, uz = cz - d * b1z;
  float n2 = fmaxf(sqrtf(ux * ux + uy * uy + uz * uz), 1e-12f);
  float b2x = ux / n2, b2y = uy / n2, b2z = uz / n2;
  float b3x = b1y * b2z - b1z * b2y;
  float b3y = b1z * b2x - b1x * b2z;
  float b3z = b1x * b2y - b1y * b2x;
  float* o = out + (size_t)r * 16;
  o[0] = b1x; o[1] = b1y; o[2] = b1z; o[3] = tx;
  o[4] = b2x; o[5] = b2y; o[6] = b2z; o[7] = ty;
  o[8] = b3x; o[9] = b3y; o[10] = b3z; o[11] = tz;
  o[12] = 0.f; o[13] = 0.f; o[14] = 0.f; o[15] = 1.f;
}

// ---------------- host-side dispatch ----------------
static void launch_fused(int variant, int grid, const FArgs& fa, hipStream_t st) {
  if (variant == 0) k_fused<128, 352, 128><<<dim3(grid), dim3(128), 0, st>>>(fa);
  else              k_fused<16, 512, 16><<<dim3(grid), dim3(128), 0, st>>>(fa);
}

static const float SA1_R[3] = {0.1f, 0.2f, 0.4f};
static const int   SA1_K[3] = {16, 32, 128};
static const int   SA1_W[3][3] = {{16, 16, 32}, {32, 32, 64}, {32, 32, 64}};
static const int   SA1_OFF[3] = {0, 32, 96};
static const float SA2_R[3] = {0.2f, 0.4f, 0.8f};
static const int   SA2_K[3] = {32, 64, 128};
static const int   SA2_W[3][3] = {{32, 32, 64}, {64, 64, 128}, {64, 64, 128}};
static const int   SA2_OFF[3] = {3, 67, 195};
static const int   SA3_W[3] = {128, 256, 512};

extern "C" void kernel_launch(void* const* d_in, const int* in_sizes, int n_in,
                              void* d_out, int out_size, void* d_ws, size_t ws_size,
                              hipStream_t stream) {
  (void)in_sizes; (void)n_in; (void)out_size; (void)ws_size;
  auto P = [&](int leaf) -> const float* { return (const float*)d_in[3 + leaf]; };
  const float* dSup = (const float*)d_in[0];
  const float* dObj = (const float*)d_in[1];
  const float* dZ   = (const float*)d_in[2];
  float* outp = (float*)d_out;

  // ---- workspace carve (floats, 256B aligned slices) ----
  float* w = (float*)d_ws;
  auto alloc = [&](size_t n) -> float* { float* p = w; w += ((n + 63) & ~(size_t)63); return p; };
  float* xyz   = alloc((size_t)B * N * 3);
  float* nx1   = alloc((size_t)B * S1 * 3);
  float* nx2   = alloc((size_t)B * S2 * 3);
  float* l1p   = alloc((size_t)B * S1 * 160);
  float* l2cat = alloc((size_t)B * S2 * 323);
  float* l3    = alloc((size_t)B * 512);
  float* fcmid = alloc((size_t)B * 256);
  float* featS = alloc((size_t)B * 128);
  float* featO = alloc((size_t)B * 128);
  float* zrows = alloc((size_t)B * MM * 3);
  float* zfeat = alloc((size_t)B * MM * 64);
  float* fusev = alloc((size_t)B * MM * 320);
  float* fm1   = alloc((size_t)B * MM * 128);
  float* fm2   = alloc((size_t)B * MM * 64);
  float* stats = alloc(1024);
  float* coef  = alloc(3072);
  h16*   hp    = (h16*)alloc(620000);      // f16 padded weights (~0.93M halves needed)

  auto convp = [&](int leaf, int Cin, int Cout) -> ConvP {
    ConvP c;
    c.b = P(leaf + 1); c.g = P(leaf + 2); c.be = P(leaf + 3);
    c.Cin = Cin; c.Cout = Cout; c.CinPad = (Cin + 31) & ~31;
    c.Wp = hp; hp += (size_t)Cout * c.CinPad;
    int tot = Cout * c.CinPad;
    k_cvtw<<<dim3((tot + 255) / 256), dim3(256), 0, stream>>>(P(leaf), c.Wp, Cin, c.CinPad, tot);
    return c;
  };

  // stats-pass ladder + output pass for an nl-layer stack
  auto runStack = [&](FArgs fa, const ConvP* cv, int nl, int grid, long Rtot,
                      int variant, int pool, float* out, int ostr, int ooff) {
    for (int i = 0; i < nl; i++) {
      fa.L[i].W = cv[i].Wp; fa.L[i].bias = cv[i].b;
      fa.L[i].Cin = cv[i].Cin; fa.L[i].CinPad = cv[i].CinPad; fa.L[i].Cout = cv[i].Cout;
      fa.L[i].acoef = coef + i * 1024; fa.L[i].ccoef = coef + i * 1024 + 512;
    }
    fa.stats = stats;
    for (int d = 1; d <= nl; d++) {
      int C = cv[d - 1].Cout;
      k_zero<<<dim3((2 * C + 255) / 256), dim3(256), 0, stream>>>(stats, 2 * C);
      FArgs s = fa; s.nl = d; s.apply_last = 0; s.do_pool = 0; s.out = nullptr;
      s.out_stride = 0; s.out_off = 0;
      launch_fused(variant, grid, s, stream);
      k_finalize<<<dim3((C + 255) / 256), dim3(256), 0, stream>>>(
          stats, cv[d - 1].g, cv[d - 1].be,
          coef + (d - 1) * 1024, coef + (d - 1) * 1024 + 512, C, 1.0f / (float)Rtot);
    }
    fa.nl = nl; fa.apply_last = 1; fa.do_pool = pool;
    fa.out = out; fa.out_stride = ostr; fa.out_off = ooff;
    launch_fused(variant, grid, fa, stream);
  };

  auto runFE = [&](const float* pc, int base, float* feat) {
    int tot = B * 3 * N;
    k_bcn_to_bnc<<<dim3((tot + 255) / 256), dim3(256), 0, stream>>>(pc, xyz, B, 3, N);
    k_fps<<<dim3(B), dim3(256), 0, stream>>>(xyz, nx1, N, S1);
    for (int i = 0; i < 3; i++) {                               // SA1 MSG
      ConvP cv[3];
      for (int j = 0; j < 3; j++)
        cv[j] = convp(base + 8 + i * 12 + j * 4, j == 0 ? 3 : SA1_W[i][j - 1], SA1_W[i][j]);
      FArgs fa{}; fa.mode = 0; fa.xyz = xyz; fa.points = nullptr; fa.newxyz = nx1;
      fa.Nn = N; fa.Ss = S1; fa.Kk = SA1_K[i]; fa.Cpts = 0; fa.Cin0 = 3;
      fa.r2 = SA1_R[i] * SA1_R[i];
      runStack(fa, cv, 3, B * S1, (long)B * S1 * SA1_K[i], 0, 1, l1p, 160, SA1_OFF[i]);
    }
    k_fps<<<dim3(B), dim3(256), 0, stream>>>(nx1, nx2, S1, S2);
    for (int i = 0; i < 3; i++) {                               // SA2 MSG
      ConvP cv[3];
      for (int j = 0; j < 3; j++)
        cv[j] = convp(base + 44 + i * 12 + j * 4, j == 0 ? 163 : SA2_W[i][j - 1], SA2_W[i][j]);
      FArgs fa{}; fa.mode = 0; fa.xyz = nx1; fa.points = l1p; fa.newxyz = nx2;
      fa.Nn = S1; fa.Ss = S2; fa.Kk = SA2_K[i]; fa.Cpts = 160; fa.Cin0 = 163;
      fa.r2 = SA2_R[i] * SA2_R[i];
      runStack(fa, cv, 3, B * S2, (long)B * S2 * SA2_K[i], 0, 1, l2cat, 323, SA2_OFF[i]);
    }
    k_copy3<<<dim3((B * S2 * 3 + 255) / 256), dim3(256), 0, stream>>>(nx2, l2cat, B * S2);
    {                                                           // SA3 (group-all)
      ConvP cv[3];
      for (int j = 0; j < 3; j++)
        cv[j] = convp(base + 80 + j * 4, j == 0 ? 323 : SA3_W[j - 1], SA3_W[j]);
      FArgs fa{}; fa.mode = 1; fa.rows = l2cat; fa.Kk = 128; fa.Cin0 = 323;
      runStack(fa, cv, 3, B, (long)B * 128, 0, 1, l3, 512, 0);
    }
    {                                                           // FE fc1 (512->256)
      ConvP cv[1] = {convp(base + 0, 512, 256)};
      FArgs fa{}; fa.mode = 1; fa.rows = l3; fa.Kk = 16; fa.Cin0 = 512;
      runStack(fa, cv, 1, 1, 16, 1, 0, fcmid, 256, 0);
    }
    {                                                           // FE fc2 (256->128)
      ConvP cv[1] = {convp(base + 4, 256, 128)};
      FArgs fa{}; fa.mode = 1; fa.rows = fcmid; fa.Kk = 16; fa.Cin0 = 256;
      runStack(fa, cv, 1, 1, 16, 1, 0, feat, 128, 0);
    }
  };

  runFE(dSup, 104, featS);     // 'sup' leaves start at 104
  runFE(dObj, 8,   featO);     // 'obj' leaves start at 8

  // ---- z MLP (3->32->64), BN over (0,1) => 512 rows ----
  {
    int tot = B * 3 * MM;
    k_bcn_to_bnc<<<dim3((tot + 255) / 256), dim3(256), 0, stream>>>(dZ, zrows, B, 3, MM);
    ConvP cv[2] = {convp(196, 3, 32), convp(200, 32, 64)};
    FArgs fa{}; fa.mode = 1; fa.rows = zrows; fa.Kk = 128; fa.Cin0 = 3;
    runStack(fa, cv, 2, 4, 512, 0, 0, zfeat, 64, 0);
  }

  // ---- fusion + FC head ----
  {
    int tot = B * MM * 320;
    k_fuse<<<dim3((tot + 255) / 256), dim3(256), 0, stream>>>(featS, featO, zfeat, fusev);
  }
  {
    ConvP cv[1] = {convp(0, 320, 128)};
    FArgs fa{}; fa.mode = 1; fa.rows = fusev; fa.Kk = 128; fa.Cin0 = 320;
    runStack(fa, cv, 1, 4, 512, 0, 0, fm1, 128, 0);
  }
  {
    ConvP cv[1] = {convp(4, 128, 64)};
    FArgs fa{}; fa.mode = 1; fa.rows = fm1; fa.Kk = 128; fa.Cin0 = 128;
    runStack(fa, cv, 1, 4, 512, 0, 0, fm2, 64, 0);
  }

  // ---- regression + rotation assembly ----
  k_reg<<<dim3(2), dim3(256), 0, stream>>>(fm2, P(100), P(101), P(102), P(103), outp, B * MM);
}
